// MaskedAttention_21595095564849
// MI455X (gfx1250) — matrix operand order repaired and checked by reading, not compiled, not run on hardware
//
#include <hip/hip_runtime.h>

// ---- CDNA5 / gfx1250 types ----
typedef __attribute__((ext_vector_type(16))) __bf16          v16bf;
typedef __attribute__((ext_vector_type(8)))  float           v8f;
typedef __attribute__((ext_vector_type(8)))  unsigned short  v8u16;
typedef __attribute__((ext_vector_type(16))) unsigned short  v16u16;

#define EMB   512
#define SEQ   4096
#define NB    2
#define NH    8
#define HD    64
#define NTOK  (NB*SEQ)     // 8192

static __device__ __forceinline__ unsigned short f2bf(float f) {
  unsigned int u = __float_as_uint(f);
  u += 0x7FFFu + ((u >> 16) & 1u);          // round-to-nearest-even
  return (unsigned short)(u >> 16);
}

static __device__ __forceinline__ v16bf mk_ab(v8u16 lo, v8u16 hi) {
  v16u16 v = __builtin_shufflevector(lo, hi, 0,1,2,3,4,5,6,7,8,9,10,11,12,13,14,15);
  return __builtin_bit_cast(v16bf, v);
}

static __device__ __forceinline__ v8f wmma_bf16(v16bf a, v16bf b, v8f c) {
  // D = A(16x32 bf16) * B(32x16 bf16) + C(16x16 f32)  ->  v_wmma_f32_16x16x32_bf16
  return __builtin_amdgcn_wmma_f32_16x16x32_bf16(false, a, false, b, (short)0, c, false, false);
}

static __device__ __forceinline__ float rmax16(float v) {
  v = fmaxf(v, __shfl_xor(v, 1, 32));
  v = fmaxf(v, __shfl_xor(v, 2, 32));
  v = fmaxf(v, __shfl_xor(v, 4, 32));
  v = fmaxf(v, __shfl_xor(v, 8, 32));
  return v;
}
static __device__ __forceinline__ float rsum16(float v) {
  v += __shfl_xor(v, 1, 32);
  v += __shfl_xor(v, 2, 32);
  v += __shfl_xor(v, 4, 32);
  v += __shfl_xor(v, 8, 32);
  return v;
}

// ---------------------------------------------------------------------------
// Kernel 1: transpose + convert the 4 weight matrices to bf16 [N][K]
// ---------------------------------------------------------------------------
__global__ __launch_bounds__(256)
void prep_weights(const float* __restrict__ wq, const float* __restrict__ wk,
                  const float* __restrict__ wv, const float* __restrict__ wo,
                  unsigned short* __restrict__ wt) {
  int idx = blockIdx.x * 256 + threadIdx.x;      // 4 * 512 * 512 threads
  int w = idx >> 18;
  int r = idx & ((1 << 18) - 1);
  int n = r >> 9;
  int k = r & 511;
  const float* src = (w == 0) ? wq : (w == 1) ? wk : (w == 2) ? wv : wo;
  wt[(size_t)w * (EMB * EMB) + (size_t)n * EMB + k] = f2bf(src[(size_t)k * EMB + n]);
}

// ---------------------------------------------------------------------------
// Kernel 2: Q/K/V projection.  One wave -> 16 tokens x 64 output dims.
//   Q,K stored bf16 [B,H,S,D];  V stored transposed bf16 [B,H,D,S].
// Loads for a whole k-step are issued as one batch (8x global_load_b128 for
// the B tiles + 4x b128 for A), THEN the 4 WMMAs run back-to-back, so load
// latency is covered by the f32->bf16 conversion ALU and by WMMA issue.
// ---------------------------------------------------------------------------
__global__ __launch_bounds__(256)
void qkv_proj(const float* __restrict__ emb, const unsigned short* __restrict__ wt,
              const float* __restrict__ bq, const float* __restrict__ bk,
              const float* __restrict__ bv,
              unsigned short* __restrict__ Q, unsigned short* __restrict__ K,
              unsigned short* __restrict__ Vt) {
  int wave = blockIdx.x * 8 + (threadIdx.x >> 5);
  int lane = threadIdx.x & 31;
  int mt = wave / 24;
  int rem = wave % 24;
  int which = rem >> 3;            // 0=Q 1=K 2=V
  int ng = rem & 7;                // group of 64 output dims
  int t0 = mt * 16;
  int mrow = lane & 15, half = lane >> 4;

  const unsigned short* W = wt + (size_t)which * (EMB * EMB);
  const float* bias = (which == 0) ? bq : (which == 1) ? bk : bv;

  v8f z = {0.f,0.f,0.f,0.f,0.f,0.f,0.f,0.f};
  v8f acc[4] = {z, z, z, z};

  const float* arow = emb + (size_t)(t0 + mrow) * EMB;
#pragma unroll 2
  for (int kk = 0; kk < EMB; kk += 32) {
    // ---- issue ALL loads for this k-step first ----
    v8u16 bt[4][2];
#pragma unroll
    for (int nt = 0; nt < 4; ++nt) {
      int n = ng * 64 + nt * 16 + mrow;
      const unsigned short* brow = W + (size_t)n * EMB + kk + half * 8;
      bt[nt][0] = *(const v8u16*)brow;
      bt[nt][1] = *(const v8u16*)(brow + 16);
    }
    // A operand: row m = lane&15, elems 0-7 -> K=half*8+0..7, 8-15 -> K=16+half*8+0..7
    float4 f0 = *(const float4*)(arow + kk + half * 8);
    float4 f1 = *(const float4*)(arow + kk + half * 8 + 4);
    float4 f2 = *(const float4*)(arow + kk + 16 + half * 8);
    float4 f3 = *(const float4*)(arow + kk + 16 + half * 8 + 4);
    // conversion ALU overlaps the outstanding B loads
    v16u16 au;
    au[0]=f2bf(f0.x);  au[1]=f2bf(f0.y);  au[2]=f2bf(f0.z);  au[3]=f2bf(f0.w);
    au[4]=f2bf(f1.x);  au[5]=f2bf(f1.y);  au[6]=f2bf(f1.z);  au[7]=f2bf(f1.w);
    au[8]=f2bf(f2.x);  au[9]=f2bf(f2.y);  au[10]=f2bf(f2.z); au[11]=f2bf(f2.w);
    au[12]=f2bf(f3.x); au[13]=f2bf(f3.y); au[14]=f2bf(f3.z); au[15]=f2bf(f3.w);
    v16bf a = __builtin_bit_cast(v16bf, au);
    // ---- 4 back-to-back WMMAs ----
#pragma unroll
    for (int nt = 0; nt < 4; ++nt)
      acc[nt] = wmma_bf16(a, mk_ab(bt[nt][0], bt[nt][1]), acc[nt]);
  }

  int b  = t0 >> 12;               // SEQ = 4096 = 2^12
  int s0 = t0 & (SEQ - 1);
#pragma unroll
  for (int nt = 0; nt < 4; ++nt) {
    int n = ng * 64 + nt * 16 + mrow;
    float bb = bias[n];
    int h = n >> 6, d = n & 63;
    if (which < 2) {
      unsigned short* out = (which == 0) ? Q : K;
      size_t base = (size_t)(b * NH + h) * SEQ * HD;
#pragma unroll
      for (int r = 0; r < 8; ++r) {
        int s = s0 + r + 8 * half;                  // C layout: row = r + 8*half
        out[base + (size_t)s * HD + d] = f2bf(acc[nt][r] + bb);
      }
    } else {
      // V transposed: rows r are consecutive tokens -> one packed 16B store
      v8u16 pk;
#pragma unroll
      for (int r = 0; r < 8; ++r) pk[r] = f2bf(acc[nt][r] + bb);
      size_t base = ((size_t)(b * NH + h) * HD + d) * SEQ + s0 + 8 * half;
      *(v8u16*)(Vt + base) = pk;
    }
  }
}

// ---------------------------------------------------------------------------
// Kernel 3: causal flash attention.  One wave per (b,h,16-query tile).
// ---------------------------------------------------------------------------
__global__ __launch_bounds__(256)
void attn_kernel(const unsigned short* __restrict__ Q, const unsigned short* __restrict__ K,
                 const unsigned short* __restrict__ Vt, unsigned short* __restrict__ attn) {
  __shared__ unsigned short plds[8][16 * 32];      // per-wave 16x32 bf16 P tile
  int wave = blockIdx.x * 8 + (threadIdx.x >> 5);
  int lane = threadIdx.x & 31;
  int qt = wave & 255;                             // S/16 = 256 query tiles
  int bh = wave >> 8;                              // b*NH + h
  int mrow = lane & 15, half = lane >> 4;
  unsigned short* P = plds[threadIdx.x >> 5];

  const unsigned short* qr = Q + ((size_t)bh * SEQ + qt * 16 + mrow) * HD;
  v16bf aQ0 = mk_ab(*(const v8u16*)(qr + half * 16),      *(const v8u16*)(qr + half * 16 + 8));
  v16bf aQ1 = mk_ab(*(const v8u16*)(qr + 32 + half * 16), *(const v8u16*)(qr + 32 + half * 16 + 8));

  v8f z = {0.f,0.f,0.f,0.f,0.f,0.f,0.f,0.f};
  v8f o[4] = {z, z, z, z};
  float mrun[8], lrun[8];
#pragma unroll
  for (int r = 0; r < 8; ++r) { mrun[r] = -1e30f; lrun[r] = 0.f; }

  const unsigned short* kbase = K  + (size_t)bh * SEQ * HD;
  const unsigned short* vtb   = Vt + (size_t)bh * HD * SEQ;
  int nkeys = qt * 16 + 16;

  for (int kc = 0; kc < nkeys; kc += 32) {
    if (kc + 32 < nkeys) {   // prefetch next chunk (global_prefetch_b8)
      __builtin_prefetch(kbase + (size_t)(kc + 32 + mrow) * HD, 0, 1);
      __builtin_prefetch(vtb + (size_t)mrow * SEQ + kc + 32, 0, 1);
    }
    // ---- batch-load both 16-key K subtiles (8x b128), then 4 WMMAs ----
    const unsigned short* kr0 = kbase + (size_t)(kc + mrow) * HD;
    const unsigned short* kr1 = kbase + (size_t)(kc + 16 + mrow) * HD;
    v8u16 k00a = *(const v8u16*)(kr0 + half * 16);
    v8u16 k00b = *(const v8u16*)(kr0 + half * 16 + 8);
    v8u16 k01a = *(const v8u16*)(kr0 + 32 + half * 16);
    v8u16 k01b = *(const v8u16*)(kr0 + 32 + half * 16 + 8);
    v8u16 k10a = *(const v8u16*)(kr1 + half * 16);
    v8u16 k10b = *(const v8u16*)(kr1 + half * 16 + 8);
    v8u16 k11a = *(const v8u16*)(kr1 + 32 + half * 16);
    v8u16 k11b = *(const v8u16*)(kr1 + 32 + half * 16 + 8);

    v8f s0v = z, s1v = z;
    s0v = wmma_bf16(aQ0, mk_ab(k00a, k00b), s0v);
    s0v = wmma_bf16(aQ1, mk_ab(k01a, k01b), s0v);
    s1v = wmma_bf16(aQ0, mk_ab(k10a, k10b), s1v);
    s1v = wmma_bf16(aQ1, mk_ab(k11a, k11b), s1v);

    // ---- V-tile loads issued early: latency hidden under softmax ALU ----
    v8u16 vtl[4][2];
#pragma unroll
    for (int nt = 0; nt < 4; ++nt) {
      const unsigned short* vr = vtb + (size_t)(nt * 16 + mrow) * SEQ + kc + half * 16;
      vtl[nt][0] = *(const v8u16*)vr;
      vtl[nt][1] = *(const v8u16*)(vr + 8);
    }

    bool domask = (kc + 31 > qt * 16);             // wave-uniform
    float pl0[8], pl1[8];
#pragma unroll
    for (int r = 0; r < 8; ++r) {
      float a0 = s0v[r] * 0.125f;                  // 1/sqrt(64)
      float a1 = s1v[r] * 0.125f;
      if (domask) {
        int qidx = qt * 16 + r + 8 * half;
        if (kc + mrow      > qidx) a0 += -1e9f;
        if (kc + 16 + mrow > qidx) a1 += -1e9f;
      }
      float mx = rmax16(fmaxf(a0, a1));
      float mn = fmaxf(mrun[r], mx);
      float al = __expf(mrun[r] - mn);
      mrun[r] = mn;
      float p0 = __expf(a0 - mn);
      float p1 = __expf(a1 - mn);
      lrun[r] = lrun[r] * al + rsum16(p0 + p1);
      o[0][r] *= al; o[1][r] *= al; o[2][r] *= al; o[3][r] *= al;
      pl0[r] = p0; pl1[r] = p1;
    }

    // ---- transpose P (C-layout -> A-operand layout) through LDS ----
#pragma unroll
    for (int r = 0; r < 8; ++r) {
      int row = r + 8 * half;
      P[row * 32 + mrow]      = f2bf(pl0[r]);
      P[row * 32 + 16 + mrow] = f2bf(pl1[r]);
    }
    asm volatile("s_wait_dscnt 0" ::: "memory");
    v16bf aP = mk_ab(*(const v8u16*)(P + mrow * 32 + half * 16),
                     *(const v8u16*)(P + mrow * 32 + half * 16 + 8));

    // ---- O += P * V : 4 back-to-back WMMAs ----
#pragma unroll
    for (int nt = 0; nt < 4; ++nt)
      o[nt] = wmma_bf16(aP, mk_ab(vtl[nt][0], vtl[nt][1]), o[nt]);
  }

  // ---- finalize and store attn [B,S,E] bf16 ----
  int b = bh >> 3, h = bh & 7;
#pragma unroll
  for (int nt = 0; nt < 4; ++nt) {
#pragma unroll
    for (int r = 0; r < 8; ++r) {
      float val = o[nt][r] / lrun[r];
      size_t tok = (size_t)b * SEQ + qt * 16 + r + 8 * half;
      attn[tok * EMB + h * 64 + nt * 16 + mrow] = f2bf(val);
    }
  }
}

// ---------------------------------------------------------------------------
// Kernel 4: out = attn * Wo^T + bo + emb   (fp32 x buffer)
// ---------------------------------------------------------------------------
__global__ __launch_bounds__(256)
void out_proj(const unsigned short* __restrict__ attn, const unsigned short* __restrict__ wto,
              const float* __restrict__ bo, const float* __restrict__ emb,
              float* __restrict__ x) {
  int wave = blockIdx.x * 8 + (threadIdx.x >> 5);
  int lane = threadIdx.x & 31;
  int mt = wave >> 3;
  int ng = wave & 7;
  int t0 = mt * 16;
  int mrow = lane & 15, half = lane >> 4;

  v8f z = {0.f,0.f,0.f,0.f,0.f,0.f,0.f,0.f};
  v8f acc[4] = {z, z, z, z};

  const unsigned short* arow = attn + (size_t)(t0 + mrow) * EMB;
#pragma unroll 2
  for (int kk = 0; kk < EMB; kk += 32) {
    // batch all loads first (A: 2x b128, B: 8x b128), then 4 WMMAs
    v8u16 a0 = *(const v8u16*)(arow + kk + half * 16);
    v8u16 a1 = *(const v8u16*)(arow + kk + half * 16 + 8);
    v8u16 bt[4][2];
#pragma unroll
    for (int nt = 0; nt < 4; ++nt) {
      int n = ng * 64 + nt * 16 + mrow;
      const unsigned short* brow = wto + (size_t)n * EMB + kk + half * 16;
      bt[nt][0] = *(const v8u16*)brow;
      bt[nt][1] = *(const v8u16*)(brow + 8);
    }
    v16bf a = mk_ab(a0, a1);
#pragma unroll
    for (int nt = 0; nt < 4; ++nt)
      acc[nt] = wmma_bf16(a, mk_ab(bt[nt][0], bt[nt][1]), acc[nt]);
  }
#pragma unroll
  for (int nt = 0; nt < 4; ++nt) {
    int n = ng * 64 + nt * 16 + mrow;
    float bb = bo[n];
#pragma unroll
    for (int r = 0; r < 8; ++r) {
      size_t t = (size_t)t0 + r + 8 * half;
      x[t * EMB + n] = acc[nt][r] + bb + emb[t * EMB + n];
    }
  }
}

// ---------------------------------------------------------------------------
// Kernel 5: LayerNorm per row (eps = 1e-5)
// ---------------------------------------------------------------------------
__global__ __launch_bounds__(256)
void layernorm(const float* __restrict__ x, const float* __restrict__ g,
               const float* __restrict__ bta, float* __restrict__ out) {
  __shared__ float ssum[256], ssq[256];
  int row = blockIdx.x;
  int tid = threadIdx.x;
  const float* xr = x + (size_t)row * EMB;
  float v0 = xr[tid], v1 = xr[tid + 256];
  ssum[tid] = v0 + v1;
  ssq[tid]  = v0 * v0 + v1 * v1;
  __syncthreads();
  for (int st = 128; st > 0; st >>= 1) {
    if (tid < st) { ssum[tid] += ssum[tid + st]; ssq[tid] += ssq[tid + st]; }
    __syncthreads();
  }
  float mu  = ssum[0] * (1.0f / EMB);
  float var = ssq[0] * (1.0f / EMB) - mu * mu;
  float rin = rsqrtf(var + 1e-5f);
  out[(size_t)row * EMB + tid]       = (v0 - mu) * rin * g[tid]       + bta[tid];
  out[(size_t)row * EMB + tid + 256] = (v1 - mu) * rin * g[tid + 256] + bta[tid + 256];
}

// ---------------------------------------------------------------------------
extern "C" void kernel_launch(void* const* d_in, const int* in_sizes, int n_in,
                              void* d_out, int out_size, void* d_ws, size_t ws_size,
                              hipStream_t stream) {
  (void)in_sizes; (void)n_in; (void)out_size; (void)ws_size;
  const float* emb  = (const float*)d_in[0];
  const float* wq   = (const float*)d_in[1];
  const float* bq   = (const float*)d_in[2];
  const float* wk   = (const float*)d_in[3];
  const float* bk   = (const float*)d_in[4];
  const float* wv   = (const float*)d_in[5];
  const float* bv   = (const float*)d_in[6];
  const float* wo   = (const float*)d_in[7];
  const float* bo   = (const float*)d_in[8];
  const float* ln_g = (const float*)d_in[9];
  const float* ln_b = (const float*)d_in[10];

  unsigned char* ws = (unsigned char*)d_ws;
  // workspace layout (bytes)
  unsigned short* wt   = (unsigned short*)(ws + 0);          //  4*512*512*2 = 2 MiB
  unsigned short* Qw   = (unsigned short*)(ws + 2097152);    //  8 MiB  [B,H,S,D] bf16
  unsigned short* Kw   = (unsigned short*)(ws + 10485760);   //  8 MiB
  unsigned short* Vt   = (unsigned short*)(ws + 18874368);   //  8 MiB  [B,H,D,S] bf16
  unsigned short* attn = (unsigned short*)(ws + 27262976);   //  8 MiB  [B,S,E] bf16
  float*          xbuf = (float*)(ws + 35651584);            // 16 MiB  [B,S,E] f32

  prep_weights<<<4096, 256, 0, stream>>>(wq, wk, wv, wo, wt);
  qkv_proj<<<1536, 256, 0, stream>>>(emb, wt, bq, bk, bv, Qw, Kw, Vt);          // 12288 waves
  attn_kernel<<<512, 256, 0, stream>>>(Qw, Kw, Vt, attn);                       //  4096 waves
  out_proj<<<512, 256, 0, stream>>>(attn, wt + 3 * (EMB * EMB), bo, emb, xbuf); //  4096 waves
  layernorm<<<NTOK, 256, 0, stream>>>(xbuf, ln_g, ln_b, (float*)d_out);
}
